// MaskedMultiHeadAttention_94489281428
// MI455X (gfx1250) — matrix operand order repaired and checked by reading, not compiled, not run on hardware
//
#include <hip/hip_runtime.h>

#define B_  4
#define S_  2048
#define D_  1024
#define H_  16
#define HD_ 64
#define M_  (B_ * S_)          // 8192 rows for the projections
#define SCALE_ 0.125f          // 1/sqrt(64)
#define BIG_NEG_ (-1e9f)

typedef __attribute__((ext_vector_type(16))) __bf16 v16bf;
typedef __attribute__((ext_vector_type(8)))  __bf16 v8bf;
typedef __attribute__((ext_vector_type(8)))  float  v8f;
typedef int v4i_ __attribute__((vector_size(16)));   // matches builtin param type

#if defined(__has_builtin)
# if __has_builtin(__builtin_amdgcn_global_load_async_to_lds_b128)
#  define USE_ASYNC_LDS 1
# endif
#endif
#ifndef USE_ASYNC_LDS
# define USE_ASYNC_LDS 0
#endif

#if USE_ASYNC_LDS
// one b128 async copy: 16 bytes global -> LDS, tracked by ASYNCcnt
// signature (from clang diagnostic): (v4i as1*, v4i as3*, imm offset, imm cpol)
#define ASYNC_CP16(gsrc, ldst)                                                 \
  __builtin_amdgcn_global_load_async_to_lds_b128(                              \
      (__attribute__((address_space(1))) v4i_*)(gsrc),                         \
      (__attribute__((address_space(3))) v4i_*)(ldst), 0, 0)
static __device__ __forceinline__ void async_wait0() {
  asm volatile("s_wait_asynccnt 0x0" ::: "memory");
}
#endif

static __device__ __forceinline__ __bf16 f2bf(float f) {
  union { float f; unsigned u; } v; v.f = f;
  unsigned r = (v.u + 0x7FFFu + ((v.u >> 16) & 1u)) >> 16;
  unsigned short s = (unsigned short)r;
  return __builtin_bit_cast(__bf16, s);
}

// ---------------------------------------------------------------- staging ---
__global__ void cvt_f32_bf16(const float* __restrict__ src, __bf16* __restrict__ dst, int n) {
  int i = blockIdx.x * blockDim.x + threadIdx.x;
  if (i < n) dst[i] = f2bf(src[i]);
}

__global__ void mask_cvt_x(const float* __restrict__ x, const unsigned char* __restrict__ mask,
                           __bf16* __restrict__ dst) {
  int i = blockIdx.x * blockDim.x + threadIdx.x;   // over B*S*D exactly
  int bs = i >> 10;                                // / D_
  float v = mask[bs] ? x[i] : 0.0f;
  dst[i] = f2bf(v);
}

// ----------------------------------------------------- LDS-blocked NT GEMM --
// Y[m,n] = sum_k X[m,k]*W[n,k] + bias[n].  Block tile 128x128, k-step 32,
// double-buffered LDS (32KB).  8 waves in 2x4 grid; each wave owns a 64x32
// tile = 4x2 WMMA tiles.  64 FLOP per global byte.
// OUTF32=false: bf16 out scattered into [B,H,S,HD].  OUTF32=true: fp32 out
// [M,D] with bias + query-mask (final projection).
template <bool OUTF32>
__global__ void __launch_bounds__(256)
gemm_bt(const __bf16* __restrict__ X, const __bf16* __restrict__ W,
        const float* __restrict__ bias, const unsigned char* __restrict__ mask,
        __bf16* __restrict__ outb, float* __restrict__ outf) {
  __shared__ alignas(32) __bf16 As[2][128][32];   // [buf][m][k]
  __shared__ alignas(32) __bf16 Bs[2][128][32];   // [buf][n][k]

  const int tid  = threadIdx.x;
  const int lane = tid & 31;
  const int wave = tid >> 5;
  const int NTB  = D_ / 128;                      // 8 block columns
  const int bm   = blockIdx.x / NTB;
  const int bn   = blockIdx.x % NTB;
  const int m0g  = bm * 128, n0g = bn * 128;
  const int waveM = wave >> 2;                    // 0..1
  const int waveN = wave & 3;                     // 0..3
  const int half = lane >> 4, l15 = lane & 15;

  // cooperative stage mapping: 256 threads cover 128 rows x 32 k (16 elems ea.)
  const int ldr = tid >> 1;
  const int ldk = (tid & 1) * 16;
  const __bf16* Ag = X + (size_t)(m0g + ldr) * D_ + ldk;
  const __bf16* Bg = W + (size_t)(n0g + ldr) * D_ + ldk;

  v8f acc[4][2] = {{{}, {}}, {{}, {}}, {{}, {}}, {{}, {}}};

  // ---- prologue: stage k-slab 0 into buffer 0
#if USE_ASYNC_LDS
  ASYNC_CP16(Ag,     &As[0][ldr][ldk]);
  ASYNC_CP16(Ag + 8, &As[0][ldr][ldk + 8]);
  ASYNC_CP16(Bg,     &Bs[0][ldr][ldk]);
  ASYNC_CP16(Bg + 8, &Bs[0][ldr][ldk + 8]);
  async_wait0();
#else
  {
    v8bf a0 = *(const v8bf*)(Ag);
    v8bf a1 = *(const v8bf*)(Ag + 8);
    v8bf b0 = *(const v8bf*)(Bg);
    v8bf b1 = *(const v8bf*)(Bg + 8);
    *(v8bf*)&As[0][ldr][ldk]     = a0;
    *(v8bf*)&As[0][ldr][ldk + 8] = a1;
    *(v8bf*)&Bs[0][ldr][ldk]     = b0;
    *(v8bf*)&Bs[0][ldr][ldk + 8] = b1;
  }
#endif
  __syncthreads();

  int cur = 0;
  for (int k0 = 0; k0 < D_; k0 += 32) {
    const int nxt = cur ^ 1;
    const bool hasNext = (k0 + 32) < D_;

    // ---- issue prefetch of next k-slab before compute
#if USE_ASYNC_LDS
    if (hasNext) {
      ASYNC_CP16(Ag + k0 + 32, &As[nxt][ldr][ldk]);
      ASYNC_CP16(Ag + k0 + 40, &As[nxt][ldr][ldk + 8]);
      ASYNC_CP16(Bg + k0 + 32, &Bs[nxt][ldr][ldk]);
      ASYNC_CP16(Bg + k0 + 40, &Bs[nxt][ldr][ldk + 8]);
    }
#else
    v8bf pa0, pa1, pb0, pb1;
    if (hasNext) {
      pa0 = *(const v8bf*)(Ag + k0 + 32);
      pa1 = *(const v8bf*)(Ag + k0 + 40);
      pb0 = *(const v8bf*)(Bg + k0 + 32);
      pb1 = *(const v8bf*)(Bg + k0 + 40);
    }
#endif

    // ---- compute from current buffer: 4 A-frags x 2 B-frags -> 8 WMMAs
    v16bf af[4], bfr[2];
#pragma unroll
    for (int mi = 0; mi < 4; ++mi) {
      int m = waveM * 64 + mi * 16 + l15;
      v8bf c0 = *(const v8bf*)&As[cur][m][half * 16];
      v8bf c1 = *(const v8bf*)&As[cur][m][half * 16 + 8];
#pragma unroll
      for (int j = 0; j < 8; ++j) { af[mi][j] = c0[j]; af[mi][j + 8] = c1[j]; }
    }
#pragma unroll
    for (int ni = 0; ni < 2; ++ni) {
      int n = waveN * 32 + ni * 16 + l15;
      bfr[ni] = *(const v16bf*)&Bs[cur][n][half * 16];
    }
#pragma unroll
    for (int mi = 0; mi < 4; ++mi)
#pragma unroll
      for (int ni = 0; ni < 2; ++ni)
        acc[mi][ni] = __builtin_amdgcn_wmma_f32_16x16x32_bf16(
            false, af[mi], false, bfr[ni], (short)0, acc[mi][ni], false, false);

    __syncthreads();
    if (hasNext) {
#if USE_ASYNC_LDS
      async_wait0();
#else
      *(v8bf*)&As[nxt][ldr][ldk]     = pa0;
      *(v8bf*)&As[nxt][ldr][ldk + 8] = pa1;
      *(v8bf*)&Bs[nxt][ldr][ldk]     = pb0;
      *(v8bf*)&Bs[nxt][ldr][ldk + 8] = pb1;
#endif
    }
    __syncthreads();
    cur = nxt;
  }

  // ---- epilogue
#pragma unroll
  for (int mi = 0; mi < 4; ++mi) {
#pragma unroll
    for (int ni = 0; ni < 2; ++ni) {
      const int n = n0g + waveN * 32 + ni * 16 + l15;
      const float bs = bias[n];
      if constexpr (OUTF32) {
#pragma unroll
        for (int r = 0; r < 8; ++r) {
          int m = m0g + waveM * 64 + mi * 16 + r + half * 8;
          float mk = mask[m] ? 1.0f : 0.0f;      // mask is [B,S] flat == m
          outf[(size_t)m * D_ + n] = (acc[mi][ni][r] + bs) * mk;
        }
      } else {
        const int h = n >> 6, hd = n & 63;
#pragma unroll
        for (int r = 0; r < 8; ++r) {
          int m = m0g + waveM * 64 + mi * 16 + r + half * 8;
          int bb = m >> 11, s = m & (S_ - 1);
          outb[(((size_t)(bb * H_ + h) * S_ + s) * HD_) + hd] = f2bf(acc[mi][ni][r] + bs);
        }
      }
    }
  }
}

// ---------------------------------------------------------- flash attention -
__global__ void __launch_bounds__(256)
flash_attn(const __bf16* __restrict__ Q, const __bf16* __restrict__ K,
           const __bf16* __restrict__ V, const unsigned char* __restrict__ mask,
           __bf16* __restrict__ Cx) {
  __shared__ alignas(32) __bf16 Kl[32][64];     // [key][hd]
  __shared__ alignas(32) __bf16 Vt[64][32];     // [hd][key] (transposed)
  __shared__ float pbuf[8][16][32];             // per-wave P staging

  const int lane = threadIdx.x & 31;
  const int wave = threadIdx.x >> 5;
  const int qblk = blockIdx.x & 15;             // 16 q-blocks of 128 rows
  const int bh   = blockIdx.x >> 4;             // b*H + h
  const int b    = bh >> 4;
  const int h    = bh & 15;
  const int q0   = qblk * 128 + wave * 16;
  const int half = lane >> 4;
  const int l15  = lane & 15;
  const size_t headBase = (size_t)bh * S_ * HD_;

  // Q fragments: two 16x32 A-slabs over HD=64
  const __bf16* qrow = Q + headBase + (size_t)(q0 + l15) * HD_;
  v16bf aq[2];
#pragma unroll
  for (int kc = 0; kc < 2; ++kc) {
    v8bf c0 = *(const v8bf*)(qrow + kc * 32 + half * 16);
    v8bf c1 = *(const v8bf*)(qrow + kc * 32 + half * 16 + 8);
#pragma unroll
    for (int j = 0; j < 8; ++j) { aq[kc][j] = c0[j]; aq[kc][j + 8] = c1[j]; }
  }

  float rowm[8], rsum[8];
  unsigned char qm[8];
  v8f acc[4] = {{}, {}, {}, {}};
#pragma unroll
  for (int r = 0; r < 8; ++r) {
    rowm[r] = -1e30f; rsum[r] = 0.0f;
    qm[r] = mask[b * S_ + q0 + r + half * 8];
  }

  for (int kb = 0; kb < S_; kb += 32) {
    __syncthreads();
    {   // cooperative stage of 32 keys: K row-major, V transposed
      int idx = threadIdx.x * 8;                // 256 threads x 8 = 2048 elems
      int key = idx >> 6, hd = idx & 63;
      v8bf kv = *(const v8bf*)(K + headBase + (size_t)(kb + key) * HD_ + hd);
      *(v8bf*)&Kl[key][hd] = kv;
      v8bf vv = *(const v8bf*)(V + headBase + (size_t)(kb + key) * HD_ + hd);
#pragma unroll
      for (int j = 0; j < 8; ++j) Vt[hd + j][key] = vv[j];
    }
    __syncthreads();

    // scores: two 16-key column tiles, K-dim 64 in 2 steps
    v8f cs[2] = {{}, {}};
#pragma unroll
    for (int tn = 0; tn < 2; ++tn) {
      int key = tn * 16 + l15;
#pragma unroll
      for (int kc = 0; kc < 2; ++kc) {
        v16bf bk = *(const v16bf*)&Kl[key][kc * 32 + half * 16];
        cs[tn] = __builtin_amdgcn_wmma_f32_16x16x32_bf16(false, aq[kc], false, bk,
                                                         (short)0, cs[tn], false, false);
      }
    }

    const unsigned char km0 = mask[b * S_ + kb + l15];
    const unsigned char km1 = mask[b * S_ + kb + 16 + l15];

    float corr[8];
#pragma unroll
    for (int r = 0; r < 8; ++r) {
      float s0 = (qm[r] && km0) ? cs[0][r] * SCALE_ : BIG_NEG_;
      float s1 = (qm[r] && km1) ? cs[1][r] * SCALE_ : BIG_NEG_;
      float tmax = fmaxf(s0, s1);
#pragma unroll
      for (int d = 1; d < 16; d <<= 1) tmax = fmaxf(tmax, __shfl_xor(tmax, d, 32));
      float mnew = fmaxf(rowm[r], tmax);
      float c  = __expf(rowm[r] - mnew);
      float p0 = __expf(s0 - mnew);
      float p1 = __expf(s1 - mnew);
      float ts = p0 + p1;
#pragma unroll
      for (int d = 1; d < 16; d <<= 1) ts += __shfl_xor(ts, d, 32);
      rowm[r] = mnew;
      rsum[r] = rsum[r] * c + ts;
      corr[r] = c;
      int row = r + half * 8;
      pbuf[wave][row][l15] = p0;
      pbuf[wave][row][16 + l15] = p1;
    }
#pragma unroll
    for (int t = 0; t < 4; ++t)
#pragma unroll
      for (int r = 0; r < 8; ++r) acc[t][r] *= corr[r];

    // same-wave LDS RAW fence before re-reading P in A-layout
    asm volatile("s_wait_dscnt 0" ::: "memory");

    v16bf ap;
#pragma unroll
    for (int j = 0; j < 8; ++j) {
      ap[j]     = f2bf(pbuf[wave][l15][half * 16 + j]);
      ap[j + 8] = f2bf(pbuf[wave][l15][half * 16 + 8 + j]);
    }
#pragma unroll
    for (int t = 0; t < 4; ++t) {
      int hd = t * 16 + l15;
      v16bf bv = *(const v16bf*)&Vt[hd][half * 16];
      acc[t] = __builtin_amdgcn_wmma_f32_16x16x32_bf16(false, ap, false, bv,
                                                       (short)0, acc[t], false, false);
    }
  }

  // normalize and merge heads -> Cx[B,S,D] bf16
#pragma unroll
  for (int r = 0; r < 8; ++r) {
    float inv = (rsum[r] > 0.0f && qm[r]) ? 1.0f / rsum[r] : 0.0f;
    int row = q0 + r + half * 8;
    __bf16* crow = Cx + (size_t)(b * S_ + row) * D_ + h * HD_;
#pragma unroll
    for (int t = 0; t < 4; ++t) crow[t * 16 + l15] = f2bf(acc[t][r] * inv);
  }
}

// ------------------------------------------------------------------ launch --
extern "C" void kernel_launch(void* const* d_in, const int* in_sizes, int n_in,
                              void* d_out, int out_size, void* d_ws, size_t ws_size,
                              hipStream_t stream) {
  const float* x  = (const float*)d_in[0];
  const unsigned char* mask = (const unsigned char*)d_in[1];
  const float* Wq = (const float*)d_in[2];
  const float* bq = (const float*)d_in[3];
  const float* Wk = (const float*)d_in[4];
  const float* bk = (const float*)d_in[5];
  const float* Wv = (const float*)d_in[6];
  const float* bv = (const float*)d_in[7];
  const float* Wo = (const float*)d_in[8];
  const float* bo = (const float*)d_in[9];
  float* out = (float*)d_out;

  const size_t NX = (size_t)M_ * D_;   // 8M elems
  const size_t NW = (size_t)D_ * D_;   // 1M elems
  __bf16* Xb  = (__bf16*)d_ws;
  __bf16* Wqb = Xb + NX;
  __bf16* Wkb = Wqb + NW;
  __bf16* Wvb = Wkb + NW;
  __bf16* Wob = Wvb + NW;
  __bf16* Qb  = Wob + NW;              // [B,H,S,HD]
  __bf16* Kb  = Qb + NX;
  __bf16* Vb  = Kb + NX;
  __bf16* Cb  = Vb + NX;               // [B,S,D] merged-head context

  dim3 blk(256);
  cvt_f32_bf16<<<(int)(NW / 256), blk, 0, stream>>>(Wq, Wqb, (int)NW);
  cvt_f32_bf16<<<(int)(NW / 256), blk, 0, stream>>>(Wk, Wkb, (int)NW);
  cvt_f32_bf16<<<(int)(NW / 256), blk, 0, stream>>>(Wv, Wvb, (int)NW);
  cvt_f32_bf16<<<(int)(NW / 256), blk, 0, stream>>>(Wo, Wob, (int)NW);
  mask_cvt_x<<<(int)(NX / 256), blk, 0, stream>>>(x, mask, Xb);

  // (M/128)*(N/128) = 64*8 = 512 workgroups per GEMM
  gemm_bt<false><<<512, blk, 0, stream>>>(Xb, Wqb, bq, nullptr, Qb, nullptr);
  gemm_bt<false><<<512, blk, 0, stream>>>(Xb, Wkb, bk, nullptr, Kb, nullptr);
  gemm_bt<false><<<512, blk, 0, stream>>>(Xb, Wvb, bv, nullptr, Vb, nullptr);

  // B*H*(S/128) = 1024 blocks
  flash_attn<<<1024, blk, 0, stream>>>(Qb, Kb, Vb, mask, Cb);

  gemm_bt<true><<<512, blk, 0, stream>>>(Cb, Wob, bo, mask, nullptr, out);
}
